// LiquidCell_82695300317264
// MI455X (gfx1250) — compile-verified, matrix operand order checked by reading, not run on hardware
//
#include <hip/hip_runtime.h>
#include <math.h>

typedef __attribute__((ext_vector_type(8)))  float        v8f;
typedef __attribute__((ext_vector_type(16))) __bf16       v16bf;
typedef __attribute__((ext_vector_type(4)))  unsigned int v4u;
typedef __attribute__((ext_vector_type(8)))  int          v8i;
typedef __attribute__((ext_vector_type(4)))  int          v4i;

#define HDIM  2048
#define BATCH 4096
#define IDIM  1024
#define TILE  64     // block C tile 64x64
#define KC    32     // K per stage = one bf16 WMMA k-depth
#define LDSW  40     // bf16 elems per LDS row: 32 data + 8 pad (TDM pads for us)
#define BDIM  128    // 4 waves, 2x2, each wave owns 32x32 of C
#define DT_C  0.1f

union Frag { v4u q[2]; v16bf v; };  // 16 bf16 = one WMMA A/B fragment per lane

__device__ inline unsigned short bf16_rne(float f) {
  unsigned int u = __float_as_uint(f);
  u += 0x7FFFu + ((u >> 16) & 1u);
  return (unsigned short)(u >> 16);
}
__device__ inline float bf16_to_f32(unsigned short s) {
  return __uint_as_float(((unsigned int)s) << 16);
}

// ---- Tensor Data Mover: DMA one 64-row x 32-col bf16 tile global->LDS.
// Pads each 64B row with 16B in LDS (pad_interval=16 DW, pad_amount=4 DW)
// -> LDS row stride 40 bf16 (conflict-free fragment reads).
__device__ inline void tdm_tile_load(unsigned int lds_byte_addr,
                                     const unsigned short* gsrc,
                                     unsigned int ld_elems) {
  unsigned long long ga = (unsigned long long)(uintptr_t)gsrc;
  v4u g0;
  g0[0] = 1u;                                   // count=1, user descriptor
  g0[1] = lds_byte_addr;                        // LDS destination
  g0[2] = (unsigned int)ga;                     // global_addr[31:0]
  g0[3] = (unsigned int)((ga >> 32) & 0x01FFFFFFull) | (2u << 30);  // addr[56:32] | type=2
  v8i g1;
  g1[0] = (int)((1u << 16)    // data_size = 2 bytes
              | (1u << 20)    // pad_enable
              | (3u << 22)    // pad_interval: 16 DWORDs (64B) between pads
              | (3u << 25));  // pad_amount:   4 DWORDs (16B)
  g1[1] = (int)(32u << 16);   // tensor_dim0 = 32 (lo16 at bits 63:48)
  g1[2] = (int)(64u << 16);   // tensor_dim0 hi = 0 | tensor_dim1 = 64
  g1[3] = (int)(32u << 16);   // tensor_dim1 hi = 0 | tile_dim0 = 32
  g1[4] = 64;                 // tile_dim1 = 64, tile_dim2 = 0
  g1[5] = (int)ld_elems;      // tensor_dim0_stride (row stride, elems)
  g1[6] = 0;
  g1[7] = 0;
  v4i z4 = {0, 0, 0, 0};
#if __clang_major__ >= 23
  v8i z8 = {0, 0, 0, 0, 0, 0, 0, 0};
  __builtin_amdgcn_tensor_load_to_lds(g0, g1, z4, z4, z8, 0);
#else
  __builtin_amdgcn_tensor_load_to_lds(g0, g1, z4, z4, 0);
#endif
}

// ---- Dual GEMM via 3xBF16 fp32 emulation: C1 = A*B1^T, C2 = A*B2^T.
// A/B operands are pre-split (hi,lo) bf16 planes, row-major [rows][ld].
// MODE 0: out1 = C1 + bias1[n]; out2 = C2            (loop-invariant phase)
// MODE 1: liquid-cell step update; also re-packs h   (recurrent phase)
template <int MODE>
__global__ __launch_bounds__(BDIM) void liquid_dual_gemm_bf16x3(
    const unsigned short* __restrict__ Ah,  const unsigned short* __restrict__ Al,  int lda,
    const unsigned short* __restrict__ B1h, const unsigned short* __restrict__ B1l, int ldb1,
    const unsigned short* __restrict__ B2h, const unsigned short* __restrict__ B2l, int ldb2,
    int K,
    const float* __restrict__ e_ic,     // MODE1: input_contrib [B,H]
    const float* __restrict__ e_xt,     // MODE1: x_tau         [B,H]
    const float* __restrict__ e_hin,    // MODE1: h_in f32      [B,H]
    const float* __restrict__ bias1,    // MODE0: W_in_b [H]
    const float* __restrict__ tau_b,    // MODE1: tau_adapt_b [H]
    const float* __restrict__ tau_base, // MODE1: tau_base [H]
    float* __restrict__ out1,           // MODE0: input_contrib ; MODE1: h_out f32
    float* __restrict__ out2,           // MODE0: x_tau ; MODE1: tau_out
    unsigned short* __restrict__ outp_hi,  // MODE1: packed h_out hi plane
    unsigned short* __restrict__ outp_lo,  // MODE1: packed h_out lo plane
    int write_tau) {
  // 2 stages x 6 planes (A_hi, A_lo, B1_hi, B1_lo, B2_hi, B2_lo), 64x40 bf16 each
  __shared__ unsigned short smem[2][6][TILE * LDSW];

  const int tid  = threadIdx.x;
  const int lane = tid & 31;
  const int wave = tid >> 5;
  const int wm   = wave >> 1;   // wave row (0..1)
  const int wn   = wave & 1;    // wave col (0..1)
  const int ln   = lane & 15;
  const int lg   = lane >> 4;   // half-wave group
  const int cm0  = blockIdx.y * TILE;  // batch-row base
  const int cn0  = blockIdx.x * TILE;  // hidden-col base

  const unsigned int lds0 = (unsigned int)(uintptr_t)&smem[0][0][0];
  const unsigned int PLANE_B = TILE * LDSW * 2;  // 5120B per plane

  auto issue_chunk = [&](int kb, int s) {
    const unsigned int base = lds0 + (unsigned int)(s * 6) * PLANE_B;
    tdm_tile_load(base + 0 * PLANE_B, Ah  + (size_t)cm0 * lda  + kb, (unsigned)lda);
    tdm_tile_load(base + 1 * PLANE_B, Al  + (size_t)cm0 * lda  + kb, (unsigned)lda);
    tdm_tile_load(base + 2 * PLANE_B, B1h + (size_t)cn0 * ldb1 + kb, (unsigned)ldb1);
    tdm_tile_load(base + 3 * PLANE_B, B1l + (size_t)cn0 * ldb1 + kb, (unsigned)ldb1);
    tdm_tile_load(base + 4 * PLANE_B, B2h + (size_t)cn0 * ldb2 + kb, (unsigned)ldb2);
    tdm_tile_load(base + 5 * PLANE_B, B2l + (size_t)cn0 * ldb2 + kb, (unsigned)ldb2);
  };

  v8f acc1[2][2] = {};
  v8f acc2[2][2] = {};

  if (wave == 0) issue_chunk(0, 0);

  const int NCH = K / KC;
  for (int ch = 0; ch < NCH; ++ch) {
    if (wave == 0) __builtin_amdgcn_s_wait_tensorcnt(0);  // chunk ch landed in LDS
    __syncthreads();  // publish to all waves; also: all waves done reading stage (ch+1)&1
    if (wave == 0 && ch + 1 < NCH) issue_chunk((ch + 1) * KC, (ch + 1) & 1);

    const unsigned short(*st)[TILE * LDSW] = smem[ch & 1];

    // Fragment layout (bf16 A 16x32 / B^T rows): lane l -> row l%16,
    // 8 elems at K = 8*(l/16) and 8 elems at K = 16 + 8*(l/16).
    Frag a_h[2], a_l[2], b1_h[2], b1_l[2], b2_h[2], b2_l[2];
#pragma unroll
    for (int g = 0; g < 2; ++g) {
      const int ar = (wm * 32 + g * 16 + ln) * LDSW + 8 * lg;
      a_h[g].q[0] = *(const v4u*)&st[0][ar];
      a_h[g].q[1] = *(const v4u*)&st[0][ar + 16];
      a_l[g].q[0] = *(const v4u*)&st[1][ar];
      a_l[g].q[1] = *(const v4u*)&st[1][ar + 16];
    }
#pragma unroll
    for (int j = 0; j < 2; ++j) {
      const int br = (wn * 32 + j * 16 + ln) * LDSW + 8 * lg;
      b1_h[j].q[0] = *(const v4u*)&st[2][br];
      b1_h[j].q[1] = *(const v4u*)&st[2][br + 16];
      b1_l[j].q[0] = *(const v4u*)&st[3][br];
      b1_l[j].q[1] = *(const v4u*)&st[3][br + 16];
      b2_h[j].q[0] = *(const v4u*)&st[4][br];
      b2_h[j].q[1] = *(const v4u*)&st[4][br + 16];
      b2_l[j].q[0] = *(const v4u*)&st[5][br];
      b2_l[j].q[1] = *(const v4u*)&st[5][br + 16];
    }

    // fp32-emulation: hi*hi + hi*lo + lo*hi, accumulated in f32
#pragma unroll
    for (int g = 0; g < 2; ++g) {
#pragma unroll
      for (int j = 0; j < 2; ++j) {
        acc1[g][j] = __builtin_amdgcn_wmma_f32_16x16x32_bf16(false, a_h[g].v, false, b1_h[j].v, (short)0, acc1[g][j], false, false);
        acc1[g][j] = __builtin_amdgcn_wmma_f32_16x16x32_bf16(false, a_h[g].v, false, b1_l[j].v, (short)0, acc1[g][j], false, false);
        acc1[g][j] = __builtin_amdgcn_wmma_f32_16x16x32_bf16(false, a_l[g].v, false, b1_h[j].v, (short)0, acc1[g][j], false, false);
        acc2[g][j] = __builtin_amdgcn_wmma_f32_16x16x32_bf16(false, a_h[g].v, false, b2_h[j].v, (short)0, acc2[g][j], false, false);
        acc2[g][j] = __builtin_amdgcn_wmma_f32_16x16x32_bf16(false, a_h[g].v, false, b2_l[j].v, (short)0, acc2[g][j], false, false);
        acc2[g][j] = __builtin_amdgcn_wmma_f32_16x16x32_bf16(false, a_l[g].v, false, b2_h[j].v, (short)0, acc2[g][j], false, false);
      }
    }
  }

  // ---- epilogue (C/D layout: vgpr r -> row r + 8*(lane/16), col lane%16) ----
#pragma unroll
  for (int g = 0; g < 2; ++g) {
#pragma unroll
    for (int j = 0; j < 2; ++j) {
      const int n  = cn0 + wn * 32 + j * 16 + ln;
      const int m0 = cm0 + wm * 32 + g * 16 + 8 * lg;
      const size_t idx0 = (size_t)m0 * HDIM + n;
      if (MODE == 0) {
        const float b = bias1[n];
#pragma unroll
        for (int r = 0; r < 8; ++r) {
          out1[idx0 + (size_t)r * HDIM] = acc1[g][j][r] + b;
          out2[idx0 + (size_t)r * HDIM] = acc2[g][j][r];
        }
      } else {
        const float tb  = tau_b[n];
        const float tba = tau_base[n];
#pragma unroll
        for (int r = 0; r < 8; ++r) {
          const size_t idx = idx0 + (size_t)r * HDIM;
          const float tl   = e_xt[idx] + acc2[g][j][r] + tb;
          const float tau  = tba * (0.5f + 1.0f / (1.0f + __expf(-tl)));
          const float act  = tanhf(acc1[g][j][r] + e_ic[idx]);
          const float hold = e_hin[idx];
          const float hn   = hold + DT_C * (act - hold) / tau;
          out1[idx] = hn;
          const unsigned short hh = bf16_rne(hn);      // re-pack h for next step
          outp_hi[idx] = hh;
          outp_lo[idx] = bf16_rne(hn - bf16_to_f32(hh));
          if (write_tau) out2[idx] = tau;
        }
      }
    }
  }
}

// ---- one-shot f32 -> (bf16 hi, bf16 lo) split ----
__global__ void pack_bf16_kernel(const float* __restrict__ in,
                                 unsigned short* __restrict__ hi,
                                 unsigned short* __restrict__ lo, long long n) {
  long long i = (long long)blockIdx.x * blockDim.x + threadIdx.x;
  const long long stride = (long long)gridDim.x * blockDim.x;
  for (; i < n; i += stride) {
    const float x = in[i];
    const unsigned short h = bf16_rne(x);
    hi[i] = h;
    lo[i] = bf16_rne(x - bf16_to_f32(h));
  }
}

extern "C" void kernel_launch(void* const* d_in, const int* in_sizes, int n_in,
                              void* d_out, int out_size, void* d_ws, size_t ws_size,
                              hipStream_t stream) {
  const float* x          = (const float*)d_in[0];  // [B, I]
  const float* hidden     = (const float*)d_in[1];  // [B, H]
  const float* W_rec      = (const float*)d_in[2];  // [H, H]
  const float* W_in_w     = (const float*)d_in[3];  // [H, I]
  const float* W_in_b     = (const float*)d_in[4];  // [H]
  const float* tau_base   = (const float*)d_in[5];  // [H]
  const float* tau_adaptw = (const float*)d_in[6];  // [H, I+H]
  const float* tau_adaptb = (const float*)d_in[7];  // [H]

  const size_t BH = (size_t)BATCH * HDIM;   // 8M
  const size_t BI = (size_t)BATCH * IDIM;   // 4M
  const size_t HH = (size_t)HDIM * HDIM;    // 4M
  const size_t HI = (size_t)HDIM * IDIM;    // 2M
  const size_t HT = (size_t)HDIM * (IDIM + HDIM);  // 6M

  // ---- workspace layout ----
  float* f = (float*)d_ws;
  float* input_contrib = f; f += BH;
  float* x_tau         = f; f += BH;
  float* hf[2];
  hf[0] = f; f += BH;
  hf[1] = f; f += BH;
  unsigned short* u = (unsigned short*)f;
  unsigned short* x_hi = u;  u += BI;
  unsigned short* x_lo = u;  u += BI;
  unsigned short* hp_hi[2]; unsigned short* hp_lo[2];
  hp_hi[0] = u; u += BH;  hp_lo[0] = u; u += BH;
  hp_hi[1] = u; u += BH;  hp_lo[1] = u; u += BH;
  unsigned short* wr_hi = u; u += HH;
  unsigned short* wr_lo = u; u += HH;
  unsigned short* wi_hi = u; u += HI;
  unsigned short* wi_lo = u; u += HI;
  unsigned short* ta_hi = u; u += HT;
  unsigned short* ta_lo = u; u += HT;

  float* h_out_final = (float*)d_out;       // [B, H]
  float* tau_out     = h_out_final + BH;    // [B, H]

  // ---- phase 0: split all loop-invariant operands (and h0) into bf16 hi/lo ----
  dim3 pgrid(2048), pblk(256);
  pack_bf16_kernel<<<pgrid, pblk, 0, stream>>>(x, x_hi, x_lo, (long long)BI);
  pack_bf16_kernel<<<pgrid, pblk, 0, stream>>>(hidden, hp_hi[0], hp_lo[0], (long long)BH);
  pack_bf16_kernel<<<pgrid, pblk, 0, stream>>>(W_rec, wr_hi, wr_lo, (long long)HH);
  pack_bf16_kernel<<<pgrid, pblk, 0, stream>>>(W_in_w, wi_hi, wi_lo, (long long)HI);
  pack_bf16_kernel<<<pgrid, pblk, 0, stream>>>(tau_adaptw, ta_hi, ta_lo, (long long)HT);

  dim3 grid(HDIM / TILE, BATCH / TILE);  // 32 x 64
  dim3 block(BDIM);

  // ---- phase 1: input_contrib = x@W_in^T + b ; x_tau = x@tau_adapt[:, :I]^T ----
  liquid_dual_gemm_bf16x3<0><<<grid, block, 0, stream>>>(
      x_hi, x_lo, IDIM, wi_hi, wi_lo, IDIM, ta_hi, ta_lo, IDIM + HDIM, IDIM,
      nullptr, nullptr, nullptr, W_in_b, nullptr, nullptr,
      input_contrib, x_tau, nullptr, nullptr, 0);

  // ---- phase 2: 5 recurrent steps; h lives as f32 + (hi,lo) bf16 planes ----
  const int NUM_STEPS = 5;
  const float* hin_f = hidden;
  for (int s = 0; s < NUM_STEPS; ++s) {
    const bool last = (s == NUM_STEPS - 1);
    const int cur = s & 1, nxt = 1 - cur;
    float* hout_f = last ? h_out_final : hf[cur];
    liquid_dual_gemm_bf16x3<1><<<grid, block, 0, stream>>>(
        hp_hi[cur], hp_lo[cur], HDIM,
        wr_hi, wr_lo, HDIM,
        ta_hi + IDIM, ta_lo + IDIM, IDIM + HDIM,  // h-part of tau_adapt
        HDIM,
        input_contrib, x_tau, hin_f,
        nullptr, tau_adaptb, tau_base,
        hout_f, tau_out, hp_hi[nxt], hp_lo[nxt], last ? 1 : 0);
    hin_f = hout_f;
  }
}
